// Spatial_Vision_Net_3719441679286
// MI455X (gfx1250) — matrix operand order repaired and checked by reading, not compiled, not run on hardware
//
#include <hip/hip_runtime.h>
#include <hip/hip_bf16.h>

// ---------------- problem constants ----------------
#define IMS     112
#define PADIMG  224            // 112 + 2*56
#define CSZ     113
#define NPOS    (CSZ*CSZ)      // 12769
#define NFEAT   192
#define BATCH   4
#define KTOT    (IMS*IMS)      // 12544
#define NTOT    (BATCH*NPOS)   // 51076
#define PADW    144            // 113 + 16 + 15
#define NCH     (BATCH*NFEAT)  // 768

typedef __attribute__((ext_vector_type(16))) __bf16        v16bf;
typedef __attribute__((ext_vector_type(8)))  float         v8f;
typedef __attribute__((ext_vector_type(4)))  unsigned int  u32x4;
typedef __attribute__((ext_vector_type(4)))  int           v4i;

union FragBF { u32x4 u[2]; v16bf v; };
union SVec   { unsigned short s[16]; u32x4 u[2]; };

#if __has_builtin(__builtin_amdgcn_global_load_async_to_lds_b128) && \
    __has_builtin(__builtin_amdgcn_s_wait_asynccnt)
#define USE_ASYNC_LDS 1
#endif

__device__ __forceinline__ unsigned short f2bf(float f) {
    unsigned int u = __float_as_uint(f);
    unsigned int r = u + 0x7FFFu + ((u >> 16) & 1u);   // round-to-nearest-even
    return (unsigned short)(r >> 16);
}

// ---------------- stage 0: per-image mean + mean-padding, fp32 -> bf16 ----------------
__global__ __launch_bounds__(256) void mean_pad_kernel(const float* __restrict__ img,
                                                       unsigned short* __restrict__ pad) {
    int b = blockIdx.x;
    __shared__ float red[256];
    const float* im = img + b * (IMS * IMS);
    float s = 0.f;
    for (int i = threadIdx.x; i < IMS * IMS; i += 256) s += im[i];
    red[threadIdx.x] = s;
    __syncthreads();
    for (int st = 128; st > 0; st >>= 1) {
        if (threadIdx.x < st) red[threadIdx.x] += red[threadIdx.x + st];
        __syncthreads();
    }
    float mean = red[0] * (1.0f / (IMS * IMS));
    unsigned short* pd = pad + b * (PADIMG * PADIMG);
    for (int i = threadIdx.x; i < PADIMG * PADIMG; i += 256) {
        int y = i / PADIMG, x = i - y * PADIMG;
        float v = mean;
        if (y >= 56 && y < 56 + IMS && x >= 56 && x < 56 + IMS)
            v = im[(y - 56) * IMS + (x - 56)];
        pd[i] = f2bf(v);
    }
}

// ---------------- stage 0b: filters fp32 -> bf16, [192][12544] ----------------
__global__ __launch_bounds__(256) void filt2bf_kernel(const float* __restrict__ filt,
                                                      unsigned short* __restrict__ fb, int n) {
    int i = blockIdx.x * 256 + threadIdx.x;
    if (i < n) fb[i] = f2bf(filt[i]);
}

// ---------------- stage 1: implicit-GEMM Gabor conv via bf16 WMMA ----------------
// A = filters [M=192][K=12544] bf16, B = im2col patches [K][N=51076] bf16 (gathered).
// Block tile: M=64 (4 filter tiles) x N=128.  8 wave32: wave w -> mtile (w&3),
// N-half (w>>2)*64; each wave issues 4 WMMAs per K-step re-using its A fragment.
__global__ __launch_bounds__(256) void conv_wmma_kernel(const unsigned short* __restrict__ filt,
                                                        const unsigned short* __restrict__ pad,
                                                        float* __restrict__ out) {
    __shared__ __align__(16) unsigned short ldsA[64 * 32];    // [m][k]
    __shared__ __align__(16) unsigned short ldsB[128 * 32];   // [n][k] (K-contiguous per n)

    const int t      = threadIdx.x;
    const int mblock = blockIdx.y;          // 0..2  (64 filters each)
    const int nblock = blockIdx.x;          // 0..399

    // ---- B-gather assignment: thread owns n_local, 16 consecutive k's ----
    const int n_local = t & 127;
    const int kk_base = (t >> 7) << 4;      // 0 or 16
    const int n       = nblock * 128 + n_local;
    const bool nvalid = (n < NTOT);
    const int  nb = nvalid ? n : 0;
    const int  b  = nb / NPOS;
    const int  p  = nb - b * NPOS;
    const int  oy = p / CSZ;
    const int  ox = p - oy * CSZ;
    const unsigned short* pimg = pad + b * (PADIMG * PADIMG) + oy * PADIMG + ox;
    // incremental (ky,kx) for k = ks + kk_base  (no divisions in the hot loop)
    int ky0 = 0, kx0 = kk_base;

    // ---- A-tile copy assignment: thread moves one 16B chunk (row t>>2, chunk t&3) ----
    const int arow = t >> 2;                // 0..63
    const int acol = (t & 3) * 8;           // element offset in k
    const unsigned short* afp = filt + (mblock * 64 + arow) * KTOT + acol;
    unsigned short* aldst = ldsA + arow * 32 + acol;

    // ---- WMMA fragment addressing (ISA 7.12.2, 16-bit layouts, wave32) ----
    const int lane   = t & 31;
    const int wave   = t >> 5;              // 0..7
    const int mloc   = wave & 3;            // local filter tile
    const int nsub64 = (wave >> 2) * 64;    // N half
    const int fm     = lane & 15;           // A row (M)
    const int ak0    = (lane >> 4) << 4;    // A: lanes 0-15 -> K 0..15; lanes 16-31 -> K 16..31
    const int bcol   = lane & 15;           // B column within 16-wide subtile
    const int bk0    = (lane >> 4) << 4;    // B: lanes 0-15 -> K 0..15; lanes 16-31 -> K 16..31

    v8f acc[4];
    #pragma unroll
    for (int s = 0; s < 4; ++s) acc[s] = (v8f){0.f,0.f,0.f,0.f,0.f,0.f,0.f,0.f};

    for (int ks = 0; ks < KTOT; ks += 32) {
        // ---- stage A tile (64x32 bf16, 4KB): async copy to LDS when available ----
#ifdef USE_ASYNC_LDS
        __builtin_amdgcn_global_load_async_to_lds_b128((v4i*)(afp + ks), (v4i*)aldst, 0, 0);
#else
        *(u32x4*)aldst = *(const u32x4*)(afp + ks);
#endif
        __builtin_prefetch(afp + ks + 64, 0, 1);   // global_prefetch for a later A chunk

        // ---- stage B tile (128x32 bf16, 8KB): register-buffered im2col gather ----
        SVec sv;
        {
            int kyj = ky0, kxj = kx0;
            #pragma unroll
            for (int j = 0; j < 16; ++j) {
                sv.s[j] = nvalid ? pimg[kyj * PADIMG + kxj] : (unsigned short)0;
                if (++kxj == IMS) { kxj = 0; ++kyj; }
            }
        }
        kx0 += 32; if (kx0 >= IMS) { kx0 -= IMS; ++ky0; }
        *(u32x4*)(ldsB + n_local * 32 + kk_base)     = sv.u[0];
        *(u32x4*)(ldsB + n_local * 32 + kk_base + 8) = sv.u[1];

#ifdef USE_ASYNC_LDS
        __builtin_amdgcn_s_wait_asynccnt(0);
#endif
        __syncthreads();

        // ---- compute: 1 A fragment, 4 B fragments, 4 WMMAs per wave ----
        FragBF fa;
        fa.u[0] = *(const u32x4*)(ldsA + (mloc * 16 + fm) * 32 + ak0);
        fa.u[1] = *(const u32x4*)(ldsA + (mloc * 16 + fm) * 32 + ak0 + 8);
        #pragma unroll
        for (int s = 0; s < 4; ++s) {
            FragBF fb;
            const unsigned short* bp = ldsB + (nsub64 + s * 16 + bcol) * 32 + bk0;
            fb.u[0] = *(const u32x4*)(bp);
            fb.u[1] = *(const u32x4*)(bp + 8);
            acc[s] = __builtin_amdgcn_wmma_f32_16x16x32_bf16(
                         false, fa.v, false, fb.v, (short)0, acc[s], false, false);
        }
        __syncthreads();
    }

    // ---- store D (C/D VGPR layout: VGPR r -> M = r + 8*(lane>=16), N = lane&15) ----
    const int mbase = (lane >> 4) * 8;
    #pragma unroll
    for (int s = 0; s < 4; ++s) {
        int ng = nblock * 128 + nsub64 + s * 16 + bcol;
        if (ng < NTOT) {
            int b2 = ng / NPOS;
            int p2 = ng - b2 * NPOS;
            #pragma unroll
            for (int r = 0; r < 8; ++r) {
                int f = (mblock * 4 + mloc) * 16 + mbase + r;
                out[(b2 * NFEAT + f) * NPOS + p2] = acc[s][r];
            }
        }
    }
}

// ---------------- stage 2: xp = a^2, reflect pad (t16,b15,l16,r15) -> [768][144][144] ----------------
__global__ __launch_bounds__(256) void square_pad_kernel(const float* __restrict__ a,
                                                         float* __restrict__ xp) {
    int idx = blockIdx.x * 256 + threadIdx.x;
    if (idx >= NCH * PADW * PADW) return;
    int ch = idx / (PADW * PADW);
    int r  = idx - ch * (PADW * PADW);
    int y  = r / PADW;
    int x  = r - y * PADW;
    int sy = y - 16; if (sy < 0) sy = -sy; if (sy > CSZ - 1) sy = 2 * (CSZ - 1) - sy;
    int sx = x - 16; if (sx < 0) sx = -sx; if (sx > CSZ - 1) sx = 2 * (CSZ - 1) - sx;
    float v = a[ch * NPOS + sy * CSZ + sx];
    xp[idx] = v * v;
}

// ---------------- stage 3: 32-tap Gaussian along x: [768][144][144] -> [768][144][113] ----------------
__global__ __launch_bounds__(256) void gauss_x_kernel(const float* __restrict__ xp,
                                                      float* __restrict__ gx) {
    __shared__ float w[32];
    if (threadIdx.x < 32) {
        float xx = -1.f + 2.f * (float)threadIdx.x / 31.f;
        w[threadIdx.x] = 0.3989422804014327f * expf(-0.5f * xx * xx);
    }
    __syncthreads();
    int idx = blockIdx.x * 256 + threadIdx.x;
    if (idx >= NCH * PADW * CSZ) return;
    int ch = idx / (PADW * CSZ);
    int r  = idx - ch * (PADW * CSZ);
    int y  = r / CSZ;
    int x0 = r - y * CSZ;
    const float* row = xp + (ch * PADW + y) * PADW + x0;
    float s = 0.f;
    #pragma unroll
    for (int j = 0; j < 32; ++j) s += row[j] * w[j];
    gx[idx] = s;
}

// ---------------- stage 4: 32-tap Gaussian along y: [768][144][113] -> t [768][113][113] ----------------
__global__ __launch_bounds__(256) void gauss_y_kernel(const float* __restrict__ gx,
                                                      float* __restrict__ t) {
    __shared__ float w[32];
    if (threadIdx.x < 32) {
        float xx = -1.f + 2.f * (float)threadIdx.x / 31.f;
        w[threadIdx.x] = 0.3989422804014327f * expf(-0.5f * xx * xx);
    }
    __syncthreads();
    int idx = blockIdx.x * 256 + threadIdx.x;
    if (idx >= NCH * NPOS) return;
    int ch = idx / NPOS;
    int r  = idx - ch * NPOS;
    int y  = r / CSZ;
    int x  = r - y * CSZ;
    const float* col = gx + (ch * PADW + y) * CSZ + x;
    float s = 0.f;
    #pragma unroll
    for (int j = 0; j < 32; ++j) s += col[j * CSZ] * w[j];
    t[idx] = s;
}

// ---------------- stage 5: 3x3 freq/orient smoothing (zero pad) + R = a^3/(C^p + Bn) ----------------
__global__ __launch_bounds__(256) void fo_smooth_final_kernel(const float* __restrict__ a,
                                                              const float* __restrict__ t,
                                                              float* __restrict__ R) {
    int idx = blockIdx.x * 256 + threadIdx.x;
    if (idx >= BATCH * NFEAT * NPOS) return;
    int bf  = idx / NPOS;
    int pos = idx - bf * NPOS;
    int b   = bf / NFEAT;
    int f   = bf - b * NFEAT;
    int fr  = f >> 4;            // f / 16
    int o   = (f >> 1) & 7;      // (f % 16) / 2
    const float w3[3] = {0.24197072451914337f, 0.3989422804014327f, 0.24197072451914337f};
    float Bn = 0.f;
    #pragma unroll
    for (int df = -1; df <= 1; ++df) {
        int fr2 = fr + df;
        if (fr2 < 0 || fr2 >= 12) continue;
        #pragma unroll
        for (int dq = -1; dq <= 1; ++dq) {
            int o2 = o + dq;
            if (o2 < 0 || o2 >= 8) continue;
            int f2 = f + df * 16 + dq * 2;
            Bn += w3[df + 1] * w3[dq + 1] * t[(b * NFEAT + f2) * NPOS + pos];
        }
    }
    float A = a[idx];
    R[idx] = (A * A * A) / (0.0625f + Bn);   // C^P = 0.25^2
}

// ---------------- host-side orchestration ----------------
extern "C" void kernel_launch(void* const* d_in, const int* in_sizes, int n_in,
                              void* d_out, int out_size, void* d_ws, size_t ws_size,
                              hipStream_t stream) {
    const float* images  = (const float*)d_in[0];   // [4,1,112,112]
    const float* filters = (const float*)d_in[1];   // [192,1,112,112]
    float* R = (float*)d_out;                       // [4,192,113,113]

    char* ws = (char*)d_ws;
    size_t o_pad  = 0;                                    // bf16 [4][224][224]
    size_t o_filt = o_pad  + (size_t)BATCH * PADIMG * PADIMG * 2;       // bf16 [192][12544]
    size_t o_a    = o_filt + (size_t)NFEAT * KTOT * 2;                  // f32  [4][192][113][113]
    size_t o_xp   = o_a    + (size_t)BATCH * NFEAT * NPOS * 4;          // f32  [768][144][144] (reused for t)
    size_t o_gx   = o_xp   + (size_t)NCH * PADW * PADW * 4;             // f32  [768][144][113]

    unsigned short* padbf  = (unsigned short*)(ws + o_pad);
    unsigned short* filtbf = (unsigned short*)(ws + o_filt);
    float* a  = (float*)(ws + o_a);
    float* xp = (float*)(ws + o_xp);
    float* gx = (float*)(ws + o_gx);
    float* t  = xp;   // gauss_y reads gx only -> safe to overwrite xp region

    mean_pad_kernel<<<BATCH, 256, 0, stream>>>(images, padbf);

    int nfilt = NFEAT * KTOT;
    filt2bf_kernel<<<(nfilt + 255) / 256, 256, 0, stream>>>(filters, filtbf, nfilt);

    dim3 cgrid((NTOT + 127) / 128, NFEAT / 64);   // 400 x 3
    conv_wmma_kernel<<<cgrid, 256, 0, stream>>>(filtbf, padbf, a);

    int nsq = NCH * PADW * PADW;
    square_pad_kernel<<<(nsq + 255) / 256, 256, 0, stream>>>(a, xp);

    int ngx = NCH * PADW * CSZ;
    gauss_x_kernel<<<(ngx + 255) / 256, 256, 0, stream>>>(xp, gx);

    int ngy = NCH * NPOS;
    gauss_y_kernel<<<(ngy + 255) / 256, 256, 0, stream>>>(gx, t);

    int nout = BATCH * NFEAT * NPOS;
    fo_smooth_final_kernel<<<(nout + 255) / 256, 256, 0, stream>>>(a, t, R);
}
